// Net_16209206575848
// MI455X (gfx1250) — matrix operand order repaired and checked by reading, not compiled, not run on hardware
//
#include <hip/hip_runtime.h>

typedef _Float16 half_t;
typedef _Float16 v8h  __attribute__((ext_vector_type(8)));
typedef _Float16 v16h __attribute__((ext_vector_type(16)));
typedef float    v4f  __attribute__((ext_vector_type(4)));
typedef float    v8f  __attribute__((ext_vector_type(8)));

#define ALPHA 0.8187307530779818f
#define BETA  0.9048374180359595f
#define THR   1.0f

#define T_STEPS 100
#define Bsz 512
#define NI  700
#define NIP 704   // NI padded to multiple of 32 (22 K-windows; last one masked)
#define NH  1024
#define NO  20
#define NOP 32    // NO padded to two 16-wide N tiles
#define LS  40    // LDS tile row stride in halves (32 data + 8 pad, bank-spread)

// ---------------------------------------------------------------------------
// Workgroup barrier that only drains DS ops (LDS visibility), leaving global
// prefetch loads (register destinations) in flight across the barrier.
// CDNA5 split barrier: s_barrier_signal -1 / s_barrier_wait -1.
// ---------------------------------------------------------------------------
__device__ __forceinline__ void wg_barrier_lds() {
    asm volatile("s_wait_dscnt 0x0\n\t"
                 "s_barrier_signal -1\n\t"
                 "s_barrier_wait -1" ::: "memory");
}

// ---------------------------------------------------------------------------
// 32-byte helpers (16B-aligned safe for LDS rows with stride 40 halves)
// ---------------------------------------------------------------------------
union H16 { v16h v; v8h h[2]; };

__device__ __forceinline__ v16h g_load32(const half_t* p) {
    H16 u;
    u.h[0] = *(const v8h*)p;
    u.h[1] = *(const v8h*)(p + 8);
    return u.v;
}
__device__ __forceinline__ void lds_store32(half_t* p, v16h v) {
    H16 u; u.v = v;
    *(v8h*)p = u.h[0];
    *(v8h*)(p + 8) = u.h[1];
}
__device__ __forceinline__ v16h lds_load32(const half_t* p) {
    H16 u;
    u.h[0] = *(const v8h*)p;
    u.h[1] = *(const v8h*)(p + 8);
    return u.v;
}

// A-fragment from an LDS tile (row-major [64][LS], K window at cols 0..31):
// lane<16: row base+lane, K=0..15 ; lane>=16: row base+lane-16, K=16..31
__device__ __forceinline__ v16h lds_a_frag(const half_t* tile, int rowBase, int lane) {
    const int hs  = lane >> 4;
    const half_t* p = tile + (rowBase + (lane & 15)) * LS + hs * 16;
    H16 u;
    u.h[0] = *(const v8h*)(p);
    u.h[1] = *(const v8h*)(p + 8);
    return u.v;
}
// B-fragment from an LDS tile: lane<16 holds K=0..15 of row nBase+lane; lane>=16 K=16..31.
__device__ __forceinline__ v16h lds_b_frag(const half_t* tile, int nBase, int lane) {
    return lds_load32(tile + (nBase + (lane & 15)) * LS + (lane >> 4) * 16);
}

// Direct-from-global fragments (used by the small layer-2 GEMM)
__device__ __forceinline__ v16h load_a_frag(const half_t* __restrict__ A, int ld,
                                            int mBase, int k, int lane) {
    const int hs  = lane >> 4;
    const half_t* p = A + (size_t)(mBase + (lane & 15)) * ld + k + hs * 16;
    H16 u;
    u.h[0] = *(const v8h*)(p);
    u.h[1] = *(const v8h*)(p + 8);
    return u.v;
}
__device__ __forceinline__ v16h load_b_frag(const half_t* __restrict__ W, int ld,
                                            int nBase, int k, int lane) {
    return g_load32(W + (size_t)(nBase + (lane & 15)) * ld + k + (lane >> 4) * 16);
}

__device__ __forceinline__ v8f wmma_f16(v16h a, v16h b, v8f c) {
    return __builtin_amdgcn_wmma_f32_16x16x32_f16(false, a, false, b, (short)0, c,
                                                  false, false);
}

__device__ __forceinline__ v16h pack_f16(v4f r0, v4f r1, v4f r2, v4f r3) {
    v16h a;
#pragma unroll
    for (int i = 0; i < 4; ++i) {
        a[i]      = (half_t)r0[i];
        a[i + 4]  = (half_t)r1[i];
        a[i + 8]  = (half_t)r2[i];
        a[i + 12] = (half_t)r3[i];
    }
    return a;
}

// ---------------------------------------------------------------------------
// One-time (per call) prep: padded f16 weight copies + state zero-init
// ---------------------------------------------------------------------------
__global__ void prep_kernel(const float* __restrict__ W1, const float* __restrict__ Wfb,
                            const float* __restrict__ W2,
                            half_t* __restrict__ W1h, half_t* __restrict__ Wfbh,
                            half_t* __restrict__ W2h, half_t* __restrict__ spk0,
                            float* __restrict__ syn1, float* __restrict__ mem1,
                            float* __restrict__ syn2, float* __restrict__ mem2) {
    const int stride = gridDim.x * blockDim.x;
    const int tid = blockIdx.x * blockDim.x + threadIdx.x;
    for (int i = tid; i < NH * NIP; i += stride) {
        int r = i / NIP, k = i % NIP;
        W1h[i] = (half_t)((k < NI) ? W1[(size_t)r * NI + k] : 0.f);
    }
    for (int i = tid; i < NH * NH; i += stride) Wfbh[i] = (half_t)Wfb[i];
    for (int i = tid; i < NOP * NH; i += stride) {
        int r = i / NH;
        W2h[i] = (half_t)((r < NO) ? W2[i] : 0.f);
    }
    for (int i = tid; i < Bsz * NH; i += stride) {
        spk0[i] = (half_t)0.f; syn1[i] = 0.f; mem1[i] = 0.f;
    }
    for (int i = tid; i < Bsz * NOP; i += stride) { syn2[i] = 0.f; mem2[i] = 0.f; }
}

// ---------------------------------------------------------------------------
// Layer 1: cur1 = x_t @ W1h^T + spkPrev @ Wfbh^T + b1 + bfb ; Synaptic update.
// Block = 128 threads (4 waves), tile 64(M) x 64(N); wave owns a 16x64 strip.
// Double-buffered LDS pipeline, depth 2: global->regs (k+2), regs->LDS (k+1),
// LDS->wmma (k). One DS-only split barrier per K window keeps global prefetch
// loads in flight across barriers. Grid (8,16) = 128 blocks.
// ---------------------------------------------------------------------------
__global__ __launch_bounds__(128) void layer1_kernel(
    const float* __restrict__ xt,        // x + t*Bsz*NI : [Bsz][NI] f32
    const half_t* __restrict__ W1h,      // [NH][NIP]
    const half_t* __restrict__ Wfbh,     // [NH][NH]
    const half_t* __restrict__ spkPrev,  // [Bsz][NH]
    half_t* __restrict__ spkNext,        // [Bsz][NH]
    float* __restrict__ syn1,            // [Bsz][NH]
    float* __restrict__ mem1,            // [Bsz][NH]
    const float* __restrict__ b1,        // [NH]
    const float* __restrict__ bfb) {     // [NH]
    __shared__ half_t lsA[2][64 * LS];
    __shared__ half_t lsB[2][64 * LS];

    const int tid  = threadIdx.x;
    const int lane = tid & 31;
    const int wave = tid >> 5;
    const int M0s  = blockIdx.x * 64;
    const int N0   = blockIdx.y * 64;
    const int Mw   = wave * 16;       // wave's strip inside the A tile
    const int sr   = tid >> 1;        // staging row 0..63
    const int sh   = tid & 1;         // staging column half
    half_t* const sA = &lsA[0][0] + sr * LS + sh * 16;  // this thread's LDS slot
    half_t* const sB = &lsB[0][0] + sr * LS + sh * 16;
    const int bufOfs = 64 * LS;       // buffer 1 offset inside lsA/lsB

    v8f acc[4];
#pragma unroll
    for (int nt = 0; nt < 4; ++nt) acc[nt] = (v8f)(0.f);

    const v4f zero4 = {0.f, 0.f, 0.f, 0.f};

    // ================= Phase 1: input projection (22 windows) =================
    {
        const float*  aB = xt  + (size_t)(M0s + sr) * NI  + sh * 16;
        const half_t* bB = W1h + (size_t)(N0  + sr) * NIP + sh * 16;

        // stage window 0 into buffer 0
        {
            v4f r0 = *(const v4f*)(aB);
            v4f r1 = *(const v4f*)(aB + 4);
            v4f r2 = *(const v4f*)(aB + 8);
            v4f r3 = *(const v4f*)(aB + 12);
            v16h bs0 = g_load32(bB);
            lds_store32(sA, pack_f16(r0, r1, r2, r3));
            lds_store32(sB, bs0);
        }
        // prefetch window 1 into registers
        v4f r0 = *(const v4f*)(aB + 32);
        v4f r1 = *(const v4f*)(aB + 36);
        v4f r2 = *(const v4f*)(aB + 40);
        v4f r3 = *(const v4f*)(aB + 44);
        v16h bs = g_load32(bB + 32);
        wg_barrier_lds();

#pragma unroll
        for (int kk = 0; kk < 22; ++kk) {
            const half_t* tA = &lsA[0][0] + (kk & 1) * bufOfs;
            const half_t* tB = &lsB[0][0] + (kk & 1) * bufOfs;
            // consume: fragments from current buffer
            v16h fa = lds_a_frag(tA, Mw, lane);
            v16h bf[4];
#pragma unroll
            for (int nt = 0; nt < 4; ++nt) bf[nt] = lds_b_frag(tB, nt * 16, lane);
            // stage window kk+1 (prefetched last iteration) into other buffer
            if (kk + 1 < 22) {
                half_t* dA = sA + ((kk + 1) & 1) * bufOfs;
                half_t* dB = sB + ((kk + 1) & 1) * bufOfs;
                lds_store32(dA, pack_f16(r0, r1, r2, r3));
                lds_store32(dB, bs);
            }
            // prefetch window kk+2 from global
            if (kk + 2 < 22) {
                const int ko = (kk + 2) * 32;
                const bool tail = (kk + 2 == 21);  // cols 700..703 -> 0 (sh==1 only)
                r0 = *(const v4f*)(aB + ko);
                r1 = *(const v4f*)(aB + ko + 4);
                r2 = *(const v4f*)(aB + ko + 8);
                r3 = (tail && sh) ? zero4 : *(const v4f*)(aB + ko + 12);
                bs = g_load32(bB + ko);
            }
#pragma unroll
            for (int nt = 0; nt < 4; ++nt) acc[nt] = wmma_f16(fa, bf[nt], acc[nt]);
            // last-iteration hazards are covered by phase 2's pre-loop barrier
            if (kk + 1 < 22) wg_barrier_lds();
        }
    }

    // ================= Phase 2: recurrent feedback (32 windows) ===============
    {
        const half_t* aB = spkPrev + (size_t)(M0s + sr) * NH + sh * 16;
        const half_t* bB = Wfbh    + (size_t)(N0  + sr) * NH + sh * 16;

        // stage window 0 into buffer 0 (phase 1's last reads were from buffer 1)
        {
            v16h as0 = g_load32(aB);
            v16h bs0 = g_load32(bB);
            lds_store32(sA, as0);
            lds_store32(sB, bs0);
        }
        // prefetch window 1
        v16h as = g_load32(aB + 32);
        v16h bs = g_load32(bB + 32);
        wg_barrier_lds();

#pragma unroll
        for (int kk = 0; kk < 32; ++kk) {
            const half_t* tA = &lsA[0][0] + (kk & 1) * bufOfs;
            const half_t* tB = &lsB[0][0] + (kk & 1) * bufOfs;
            v16h fa = lds_a_frag(tA, Mw, lane);
            v16h bf[4];
#pragma unroll
            for (int nt = 0; nt < 4; ++nt) bf[nt] = lds_b_frag(tB, nt * 16, lane);
            if (kk + 1 < 32) {
                half_t* dA = sA + ((kk + 1) & 1) * bufOfs;
                half_t* dB = sB + ((kk + 1) & 1) * bufOfs;
                lds_store32(dA, as);
                lds_store32(dB, bs);
            }
            if (kk + 2 < 32) {
                const int ko = (kk + 2) * 32;
                as = g_load32(aB + ko);
                bs = g_load32(bB + ko);
            }
#pragma unroll
            for (int nt = 0; nt < 4; ++nt) acc[nt] = wmma_f16(fa, bf[nt], acc[nt]);
            // no barrier after the final window: epilogue touches only registers
            // and global memory
            if (kk + 1 < 32) wg_barrier_lds();
        }
    }

    // ---- Epilogue: bias + Synaptic (reset from pre-update mem) ----
    const int mrow = (lane >> 4) * 8;  // C layout: lanes 16-31 hold M = 8..15
    const int ncol = lane & 15;
#pragma unroll
    for (int nt = 0; nt < 4; ++nt) {
        const int n = N0 + nt * 16 + ncol;
        const float bias = b1[n] + bfb[n];
#pragma unroll
        for (int i = 0; i < 8; ++i) {
            const int m = M0s + Mw + mrow + i;
            const size_t idx = (size_t)m * NH + n;
            const float cur = acc[nt][i] + bias;
            float s = syn1[idx];
            float u = mem1[idx];
            const float reset = (u > THR) ? 1.f : 0.f;
            s = ALPHA * s + cur;
            u = BETA * u + s - reset * THR;
            syn1[idx] = s;
            mem1[idx] = u;
            spkNext[idx] = (half_t)((u > THR) ? 1.f : 0.f);
        }
    }
}

// ---------------------------------------------------------------------------
// Layer 2: cur2 = spk1 @ W2h^T + b2 (N padded to 32); Synaptic; write outputs.
// Split-K x4: block owns one 16-row M strip; wave w accumulates K slice
// [w*256, w*256+256); partials reduced through LDS; wave 0 does the epilogue.
// Grid = 32 blocks x 128 threads.
// ---------------------------------------------------------------------------
__global__ __launch_bounds__(128) void layer2_kernel(
    const half_t* __restrict__ spk1,  // [Bsz][NH]
    const half_t* __restrict__ W2h,   // [NOP][NH]
    float* __restrict__ syn2,         // [Bsz][NOP]
    float* __restrict__ mem2,         // [Bsz][NOP]
    const float* __restrict__ b2,     // [NO]
    float* __restrict__ out, int t) {
    __shared__ float red[3][2][8][32];

    const int lane = threadIdx.x & 31;
    const int wave = threadIdx.x >> 5;
    const int M0 = blockIdx.x * 16;
    const int k0 = wave * 256;

    v8f acc[2];
    acc[0] = (v8f)(0.f);
    acc[1] = (v8f)(0.f);
#pragma unroll
    for (int kk = 0; kk < 8; ++kk) {
        const int k = k0 + kk * 32;
        v16h fa  = load_a_frag(spk1, NH, M0, k, lane);
        v16h bf0 = load_b_frag(W2h, NH, 0, k, lane);
        v16h bf1 = load_b_frag(W2h, NH, 16, k, lane);
        acc[0] = wmma_f16(fa, bf0, acc[0]);
        acc[1] = wmma_f16(fa, bf1, acc[1]);
    }

    if (wave > 0) {
#pragma unroll
        for (int nt = 0; nt < 2; ++nt)
#pragma unroll
            for (int i = 0; i < 8; ++i) red[wave - 1][nt][i][lane] = acc[nt][i];
    }
    __syncthreads();
    if (wave != 0) return;

#pragma unroll
    for (int w = 0; w < 3; ++w)
#pragma unroll
        for (int nt = 0; nt < 2; ++nt)
#pragma unroll
            for (int i = 0; i < 8; ++i) acc[nt][i] += red[w][nt][i][lane];

    const int mrow = (lane >> 4) * 8;
    const int ncol = lane & 15;
#pragma unroll
    for (int nt = 0; nt < 2; ++nt) {
        const int n = nt * 16 + ncol;
        if (n >= NO) continue;
        const float bias = b2[n];
#pragma unroll
        for (int i = 0; i < 8; ++i) {
            const int m = M0 + mrow + i;
            const size_t idx = (size_t)m * NOP + n;
            const float cur = acc[nt][i] + bias;
            float s = syn2[idx];
            float u = mem2[idx];
            const float reset = (u > THR) ? 1.f : 0.f;
            s = ALPHA * s + cur;
            u = BETA * u + s - reset * THR;
            syn2[idx] = s;
            mem2[idx] = u;
            const size_t o = (size_t)t * Bsz * NO + (size_t)m * NO + n;
            out[o] = (u > THR) ? 1.f : 0.f;           // spk2_rec
            out[(size_t)T_STEPS * Bsz * NO + o] = u;  // mem2_rec
        }
    }
}

// ---------------------------------------------------------------------------
extern "C" void kernel_launch(void* const* d_in, const int* in_sizes, int n_in,
                              void* d_out, int out_size, void* d_ws, size_t ws_size,
                              hipStream_t stream) {
    (void)in_sizes; (void)n_in; (void)out_size; (void)ws_size;
    const float* x   = (const float*)d_in[0];  // [100][512][700]
    const float* W1  = (const float*)d_in[1];  // [1024][700]
    const float* b1  = (const float*)d_in[2];  // [1024]
    const float* Wfb = (const float*)d_in[3];  // [1024][1024]
    const float* bfb = (const float*)d_in[4];  // [1024]
    const float* W2  = (const float*)d_in[5];  // [20][1024]
    const float* b2  = (const float*)d_in[6];  // [20]
    float* out = (float*)d_out;

    char* ws = (char*)d_ws;
    size_t off = 0;
    auto carve = [&](size_t bytes) -> char* {
        char* p = ws + off;
        off = (off + bytes + 255) & ~(size_t)255;
        return p;
    };
    half_t* W1h  = (half_t*)carve((size_t)NH * NIP * sizeof(half_t));
    half_t* Wfbh = (half_t*)carve((size_t)NH * NH * sizeof(half_t));
    half_t* W2h  = (half_t*)carve((size_t)NOP * NH * sizeof(half_t));
    half_t* spkA = (half_t*)carve((size_t)Bsz * NH * sizeof(half_t));
    half_t* spkB = (half_t*)carve((size_t)Bsz * NH * sizeof(half_t));
    float*  syn1 = (float*)carve((size_t)Bsz * NH * sizeof(float));
    float*  mem1 = (float*)carve((size_t)Bsz * NH * sizeof(float));
    float*  syn2 = (float*)carve((size_t)Bsz * NOP * sizeof(float));
    float*  mem2 = (float*)carve((size_t)Bsz * NOP * sizeof(float));

    prep_kernel<<<1024, 256, 0, stream>>>(W1, Wfb, W2, W1h, Wfbh, W2h,
                                          spkA, syn1, mem1, syn2, mem2);

    for (int t = 0; t < T_STEPS; ++t) {
        const half_t* prev = (t & 1) ? spkB : spkA;
        half_t* next       = (t & 1) ? spkA : spkB;
        layer1_kernel<<<dim3(8, 16), 128, 0, stream>>>(
            x + (size_t)t * Bsz * NI, W1h, Wfbh, prev, next, syn1, mem1, b1, bfb);
        layer2_kernel<<<32, 128, 0, stream>>>(next, W2h, syn2, mem2, b2, out, t);
    }
}
